// LSTMPredictor_69698729280011
// MI455X (gfx1250) — compile-verified, matrix operand order checked by reading, not run on hardware
//
#include <hip/hip_runtime.h>
#include <hip/hip_bf16.h>

// ---------------------------------------------------------------------------
// 2-layer LSTM (T=512, B=256, IN=64, H=512, OUT=64) on gfx1250 WMMA (wave32).
//
//  * One-time per call: pack weights to f16 (W0cat=[4H][IN+H], W1cat=[4H][2H],
//    Wlin=[OUT][H]), combine biases, convert x to f16, zero state.
//  * Per step t (3 kernels, stream-ordered for the recurrence):
//      A: gates0 = [x_t | h0] @ W0cat^T  -> cell -> h0', c0   (K=576)
//      B: gates1 = [h0'| h1] @ W1cat^T  -> cell -> h1', c1    (K=1024)
//      C: out_t  = h1' @ Wlin^T + b_lin -> d_out[:, t*64:(t+1)*64]
//  * Each wave owns TWO 16-row M-tiles x one 16-col N-tile x 4 gates
//    (i,f,g,o at n, n+512, n+1024, n+1536): 8 accumulators, 8 WMMAs per
//    K-chunk sharing 4 B fragments -> high in-wave ILP to cover load latency
//    (grid is too small for occupancy-based hiding on a big CDNA5 part).
// ---------------------------------------------------------------------------

#define T_STEPS 512
#define BATCH   256
#define IN_DIM  64
#define HID     512
#define OUT_DIM 64
#define K0TOT   (IN_DIM + HID)   // 576
#define K1TOT   (2 * HID)        // 1024

typedef __attribute__((ext_vector_type(16))) _Float16 v16h;
typedef __attribute__((ext_vector_type(8)))  _Float16 v8h;
typedef __attribute__((ext_vector_type(8)))  float    v8f;

#define WMMA_F16(A, B, C) \
  __builtin_amdgcn_wmma_f32_16x16x32_f16(false, (A), false, (B), (short)0, (C), false, false)

// ---- math helpers ---------------------------------------------------------
__device__ __forceinline__ float fsig(float x) {
  return 1.0f / (1.0f + __expf(-x));
}
__device__ __forceinline__ float ftanh(float x) {
  float a = fabsf(x);
  float e = __expf(-2.0f * a);
  float t = (1.0f - e) / (1.0f + e);
  return copysignf(t, x);
}

// ---- WMMA fragment load (16-bit A/B layout, wave32) -----------------------
// Lane L (L<16): row = row0+L,    K = k0+[0..7] and k0+16..k0+23
// Lane L (>=16): row = row0+L-16, K = k0+8..k0+15 and k0+24..k0+31
// Two 16-byte loads per lane; ld = row stride in elements (16B aligned).
__device__ __forceinline__ v16h load_frag(const _Float16* __restrict__ base,
                                          int ld, int row0, int k0, int lane) {
  const int r  = row0 + (lane & 15);
  const int kb = k0 + ((lane >> 4) << 3);
  const _Float16* p = base + r * ld + kb;
  v8h lo = *(const v8h*)(p);
  v8h hi = *(const v8h*)(p + 16);
  v16h out;
#pragma unroll
  for (int j = 0; j < 8; ++j) { out[j] = lo[j]; out[j + 8] = hi[j]; }
  return out;
}

// ---------------------------------------------------------------------------
// Setup kernels (run once per call; cheap, fully parallel)
// ---------------------------------------------------------------------------
__global__ void pack_w0_kernel(const float* __restrict__ Wih0,
                               const float* __restrict__ Whh0,
                               _Float16* __restrict__ W0) {
  int idx = blockIdx.x * 256 + threadIdx.x;            // 2048*576
  if (idx >= 4 * HID * K0TOT) return;
  int g = idx / K0TOT, k = idx - g * K0TOT;
  float v = (k < IN_DIM) ? Wih0[g * IN_DIM + k] : Whh0[g * HID + (k - IN_DIM)];
  W0[idx] = (_Float16)v;
}

__global__ void pack_w1_kernel(const float* __restrict__ Wih1,
                               const float* __restrict__ Whh1,
                               _Float16* __restrict__ W1) {
  int idx = blockIdx.x * 256 + threadIdx.x;            // 2048*1024
  if (idx >= 4 * HID * K1TOT) return;
  int g = idx >> 10, k = idx & 1023;
  float v = (k < HID) ? Wih1[g * HID + k] : Whh1[g * HID + (k - HID)];
  W1[idx] = (_Float16)v;
}

__global__ void pack_wlin_kernel(const float* __restrict__ Wl,
                                 _Float16* __restrict__ WlH) {
  int idx = blockIdx.x * 256 + threadIdx.x;            // 64*512
  if (idx >= OUT_DIM * HID) return;
  WlH[idx] = (_Float16)Wl[idx];
}

__global__ void pack_bias_kernel(const float* __restrict__ bi0,
                                 const float* __restrict__ bh0,
                                 const float* __restrict__ bi1,
                                 const float* __restrict__ bh1,
                                 float* __restrict__ bias0,
                                 float* __restrict__ bias1) {
  int idx = blockIdx.x * 256 + threadIdx.x;            // 2048
  if (idx >= 4 * HID) return;
  bias0[idx] = bi0[idx] + bh0[idx];
  bias1[idx] = bi1[idx] + bh1[idx];
}

__global__ void conv_x_kernel(const float* __restrict__ x,
                              _Float16* __restrict__ x16, int n) {
  int idx = blockIdx.x * 256 + threadIdx.x;            // 512*256*64
  if (idx >= n) return;
  x16[idx] = (_Float16)x[idx];
}

__global__ void zero_state_kernel(_Float16* h0A, _Float16* h0B,
                                  _Float16* h1A, _Float16* h1B,
                                  float* c0, float* c1) {
  int idx = blockIdx.x * 256 + threadIdx.x;            // 256*512
  if (idx >= BATCH * HID) return;
  _Float16 z = (_Float16)0.0f;
  h0A[idx] = z; h0B[idx] = z; h1A[idx] = z; h1B[idx] = z;
  c0[idx] = 0.0f; c1[idx] = 0.0f;
}

// ---------------------------------------------------------------------------
// Fused gates-GEMM + LSTM cell. Block = 128 threads = 4 waves. Each wave:
// rows [m0, m0+32) as two 16-row tiles, cols [n0, n0+16) for all 4 gates.
// Grid (HID/16, BATCH/128) = (32, 2). No divergence -> EXEC all-ones at WMMA.
// ---------------------------------------------------------------------------
__device__ __forceinline__ void lstm_gates_body(
    const _Float16* __restrict__ aSrc0,  // first K section  [256][lda0]
    const _Float16* __restrict__ aSrc1,  // second K section [256][512]
    int lda0, int ksplit, int ktot,
    float* __restrict__ cSt,             // [256][512] cell state (in place)
    const _Float16* __restrict__ W,      // [2048][ktot]
    const float* __restrict__ bias,      // [2048]
    _Float16* __restrict__ hOut) {       // [256][512]
  const int wave = threadIdx.x >> 5;
  const int lane = threadIdx.x & 31;
  const int m0 = blockIdx.y * 128 + wave * 32;
  const int n0 = blockIdx.x * 16;

  v8f acc[4][2];
#pragma unroll
  for (int g = 0; g < 4; ++g) { acc[g][0] = (v8f){}; acc[g][1] = (v8f){}; }

  for (int k0 = 0; k0 < ktot; k0 += 32) {
    v16h a0, a1;
    if (k0 < ksplit) {
      a0 = load_frag(aSrc0, lda0, m0, k0, lane);
      a1 = load_frag(aSrc0, lda0, m0 + 16, k0, lane);
    } else {
      a0 = load_frag(aSrc1, HID, m0, k0 - ksplit, lane);
      a1 = load_frag(aSrc1, HID, m0 + 16, k0 - ksplit, lane);
    }
    v16h b0 = load_frag(W, ktot, 0 * HID + n0, k0, lane);
    v16h b1 = load_frag(W, ktot, 1 * HID + n0, k0, lane);
    v16h b2 = load_frag(W, ktot, 2 * HID + n0, k0, lane);
    v16h b3 = load_frag(W, ktot, 3 * HID + n0, k0, lane);
    acc[0][0] = WMMA_F16(a0, b0, acc[0][0]);
    acc[0][1] = WMMA_F16(a1, b0, acc[0][1]);
    acc[1][0] = WMMA_F16(a0, b1, acc[1][0]);
    acc[1][1] = WMMA_F16(a1, b1, acc[1][1]);
    acc[2][0] = WMMA_F16(a0, b2, acc[2][0]);
    acc[2][1] = WMMA_F16(a1, b2, acc[2][1]);
    acc[3][0] = WMMA_F16(a0, b3, acc[3][0]);
    acc[3][1] = WMMA_F16(a1, b3, acc[3][1]);
  }

  // LSTM cell, in-register (gate order i,f,g,o)
  const int n = n0 + (lane & 15);
  const float bi = bias[n], bf = bias[n + HID];
  const float bg = bias[n + 2 * HID], bo = bias[n + 3 * HID];
#pragma unroll
  for (int half = 0; half < 2; ++half) {
#pragma unroll
    for (int r = 0; r < 8; ++r) {
      const int m = m0 + half * 16 + r + ((lane >> 4) << 3);
      const float iv = fsig(acc[0][half][r] + bi);
      const float fv = fsig(acc[1][half][r] + bf);
      const float gv = ftanh(acc[2][half][r] + bg);
      const float ov = fsig(acc[3][half][r] + bo);
      const float cn = fv * cSt[m * HID + n] + iv * gv;
      cSt[m * HID + n] = cn;
      hOut[m * HID + n] = (_Float16)(ov * ftanh(cn));
    }
  }
}

__global__ void lstm_step0_kernel(const _Float16* __restrict__ x16t,  // [256][64]
                                  const _Float16* __restrict__ h0_in, // [256][512]
                                  float* __restrict__ c0,
                                  const _Float16* __restrict__ W0,    // [2048][576]
                                  const float* __restrict__ bias0,
                                  _Float16* __restrict__ h0_out) {
  lstm_gates_body(x16t, h0_in, IN_DIM, IN_DIM, K0TOT, c0, W0, bias0, h0_out);
}

__global__ void lstm_step1_kernel(const _Float16* __restrict__ h0_new,
                                  const _Float16* __restrict__ h1_in,
                                  float* __restrict__ c1,
                                  const _Float16* __restrict__ W1,    // [2048][1024]
                                  const float* __restrict__ bias1,
                                  _Float16* __restrict__ h1_out) {
  lstm_gates_body(h0_new, h1_in, HID, HID, K1TOT, c1, W1, bias1, h1_out);
}

// out_t[m][n] = sum_k h1[m][k] * Wlin[n][k] + b_lin[n], written to
// d_out[m * (T*64) + t*64 + n]  (outp already offset by t*64).
// Grid (OUT/16, BATCH/128) = (4, 2); wave: two 16-row tiles.
__global__ void out_proj_kernel(const _Float16* __restrict__ h1,   // [256][512]
                                const _Float16* __restrict__ Wlin, // [64][512]
                                const float* __restrict__ b_lin,   // [64]
                                float* __restrict__ outp) {
  const int wave = threadIdx.x >> 5;
  const int lane = threadIdx.x & 31;
  const int m0 = blockIdx.y * 128 + wave * 32;
  const int n0 = blockIdx.x * 16;

  v8f acc0 = {}, acc1 = {};
#pragma unroll
  for (int kc = 0; kc < HID / 32; ++kc) {              // 16 chunks
    const int k0 = kc * 32;
    v16h a0 = load_frag(h1, HID, m0, k0, lane);
    v16h a1 = load_frag(h1, HID, m0 + 16, k0, lane);
    v16h b  = load_frag(Wlin, HID, n0, k0, lane);
    acc0 = WMMA_F16(a0, b, acc0);
    acc1 = WMMA_F16(a1, b, acc1);
  }
  const int n = n0 + (lane & 15);
  const float bb = b_lin[n];
#pragma unroll
  for (int r = 0; r < 8; ++r) {
    const int m = m0 + r + ((lane >> 4) << 3);
    outp[(size_t)m * (T_STEPS * OUT_DIM) + n] = acc0[r] + bb;
    outp[(size_t)(m + 16) * (T_STEPS * OUT_DIM) + n] = acc1[r] + bb;
  }
}

// ---------------------------------------------------------------------------
extern "C" void kernel_launch(void* const* d_in, const int* in_sizes, int n_in,
                              void* d_out, int out_size, void* d_ws, size_t ws_size,
                              hipStream_t stream) {
  const float* x     = (const float*)d_in[0];
  const float* Wih0  = (const float*)d_in[1];
  const float* Whh0  = (const float*)d_in[2];
  const float* bih0  = (const float*)d_in[3];
  const float* bhh0  = (const float*)d_in[4];
  const float* Wih1  = (const float*)d_in[5];
  const float* Whh1  = (const float*)d_in[6];
  const float* bih1  = (const float*)d_in[7];
  const float* bhh1  = (const float*)d_in[8];
  const float* Wl    = (const float*)d_in[9];
  const float* bl    = (const float*)d_in[10];
  float* out = (float*)d_out;
  (void)in_sizes; (void)n_in; (void)out_size; (void)ws_size;  // future == 0

  // Workspace carve-up (all 256B aligned); total ~25 MB, lives in L2.
  char* w = (char*)d_ws;
  size_t off = 0;
  auto carve = [&](size_t bytes) -> void* {
    void* p = (void*)(w + off);
    off += (bytes + 255) & ~(size_t)255;
    return p;
  };
  _Float16* W0   = (_Float16*)carve((size_t)4 * HID * K0TOT * 2);  // 2.25 MB
  _Float16* W1   = (_Float16*)carve((size_t)4 * HID * K1TOT * 2);  // 4 MB
  _Float16* WlH  = (_Float16*)carve((size_t)OUT_DIM * HID * 2);    // 64 KB
  float*    b0   = (float*)carve((size_t)4 * HID * 4);
  float*    b1   = (float*)carve((size_t)4 * HID * 4);
  _Float16* x16  = (_Float16*)carve((size_t)T_STEPS * BATCH * IN_DIM * 2); // 16 MB
  _Float16* h0A  = (_Float16*)carve((size_t)BATCH * HID * 2);
  _Float16* h0B  = (_Float16*)carve((size_t)BATCH * HID * 2);
  _Float16* h1A  = (_Float16*)carve((size_t)BATCH * HID * 2);
  _Float16* h1B  = (_Float16*)carve((size_t)BATCH * HID * 2);
  float*    c0   = (float*)carve((size_t)BATCH * HID * 4);
  float*    c1   = (float*)carve((size_t)BATCH * HID * 4);

  // ---- one-time setup (per call; deterministic) ----
  pack_w0_kernel<<<(4 * HID * K0TOT + 255) / 256, 256, 0, stream>>>(Wih0, Whh0, W0);
  pack_w1_kernel<<<(4 * HID * K1TOT + 255) / 256, 256, 0, stream>>>(Wih1, Whh1, W1);
  pack_wlin_kernel<<<(OUT_DIM * HID + 255) / 256, 256, 0, stream>>>(Wl, WlH);
  pack_bias_kernel<<<(4 * HID + 255) / 256, 256, 0, stream>>>(bih0, bhh0, bih1, bhh1, b0, b1);
  const int nx = T_STEPS * BATCH * IN_DIM;
  conv_x_kernel<<<(nx + 255) / 256, 256, 0, stream>>>(x, x16, nx);
  zero_state_kernel<<<(BATCH * HID + 255) / 256, 256, 0, stream>>>(h0A, h0B, h1A, h1B, c0, c1);

  // ---- recurrence: 512 serial steps, stream-ordered ----
  const dim3 gridG(HID / 16, BATCH / 128);     // 32 x 2
  const dim3 gridO(OUT_DIM / 16, BATCH / 128); // 4 x 2
  for (int t = 0; t < T_STEPS; ++t) {
    const _Float16* h0_in  = (t & 1) ? h0B : h0A;
    _Float16*       h0_out = (t & 1) ? h0A : h0B;
    const _Float16* h1_in  = (t & 1) ? h1B : h1A;
    _Float16*       h1_out = (t & 1) ? h1A : h1B;

    lstm_step0_kernel<<<gridG, 128, 0, stream>>>(
        x16 + (size_t)t * BATCH * IN_DIM, h0_in, c0, W0, b0, h0_out);
    lstm_step1_kernel<<<gridG, 128, 0, stream>>>(
        h0_out, h1_in, c1, W1, b1, h1_out);
    out_proj_kernel<<<gridO, 128, 0, stream>>>(
        h1_out, WlH, bl, out + (size_t)t * OUT_DIM);
  }
}
